// MMPM_Net_44375602102718
// MI455X (gfx1250) — matrix operand order repaired and checked by reading, not compiled, not run on hardware
//
#include <hip/hip_runtime.h>

// ---------------------------------------------------------------------------
// MMPM-Net for MI455X (gfx1250): bf16 WMMA implicit-GEMM for the 64->64 conv
// layers (the ~226 GFLOP hot path), VALU kernels for everything else.
// Round 2: register-block the WMMA kernel along M (4 pixel-tiles per wave) so
// each B fragment feeds 4 WMMAs and per-chunk address math is amortized.
// ---------------------------------------------------------------------------

typedef __attribute__((ext_vector_type(16))) __bf16          v16bf;
typedef __attribute__((ext_vector_type(8)))  float           v8f;
typedef __attribute__((ext_vector_type(8)))  unsigned short  ushort8;

#define IMG_H   160
#define IMG_W   160
#define IMG_N   4
#define NPIX    (IMG_N * IMG_H * IMG_W)   // 102400
#define NQ      8
#define CH      64
#define KCHUNKS 18                        // 9*64 / 32
#define MTILES  4                         // pixel tiles (of 16) per wave
#define WPK_PER_LAYER (KCHUNKS * 4 * 32 * 16)  // 36864
#define EPS_F   1e-08f

__device__ __forceinline__ unsigned short f32_to_bf16(float f) {
  unsigned int u = __float_as_uint(f);
  unsigned int r = (u + 0x7FFFu + ((u >> 16) & 1u)) >> 16;   // RNE
  return (unsigned short)r;
}
__device__ __forceinline__ float bf16_to_f32(unsigned short h) {
  return __uint_as_float(((unsigned int)h) << 16);
}
__device__ __forceinline__ float chmax(int c) {
  return c == 0 ? 3.f : (c == 1 ? 6.f : 50.f);
}

// ---------------- weight repack: (K,64) f32 -> B-fragment bf16 layout --------
// packed idx = ((c*4 + ntile)*32 + lane)*16 + e ;  N = ntile*16 + (lane&15),
// K = c*32 + (e>>3)*16 + (lane>>4)*8 + (e&7)   (lanes 0-15 hold K 0..7,16..23; lanes 16-31 K 8..15,24..31)
__global__ void k_repack(const float* __restrict__ w1, const float* __restrict__ w2,
                         const float* __restrict__ w3, const float* __restrict__ w4,
                         const float* __restrict__ w5, unsigned short* __restrict__ wpk) {
  int t = blockIdx.x * blockDim.x + threadIdx.x;
  if (t >= 5 * WPK_PER_LAYER) return;
  int layer = t / WPK_PER_LAYER;
  int idx   = t % WPK_PER_LAYER;
  int e     = idx & 15;
  int lane  = (idx >> 4) & 31;
  int tnt   = (idx >> 9) & 3;
  int c     = idx >> 11;
  int N     = tnt * 16 + (lane & 15);
  int K     = c * 32 + ((e >> 3) << 4) + ((lane >> 4) << 3) + (e & 7);
  const float* w = (layer == 0) ? w1 : (layer == 1) ? w2 : (layer == 2) ? w3
                 : (layer == 3) ? w4 : w5;
  wpk[t] = f32_to_bf16(w[(size_t)K * CH + N]);
}

// ---------------- init: x = clip([maxabs, 2maxabs, 1]) ; z = x ; beta = 0 ----
__global__ void k_init(const float* __restrict__ b, float* __restrict__ x,
                       float* __restrict__ z, float* __restrict__ beta) {
  int p = blockIdx.x * blockDim.x + threadIdx.x;
  if (p >= NPIX) return;
  const float* bp = b + (size_t)p * NQ;
  float A = 0.f;
  #pragma unroll
  for (int q = 0; q < NQ; ++q) A = fmaxf(A, fabsf(bp[q]));
  float x0 = fminf(fmaxf(A, 0.f), 3.f);
  float x1 = fminf(fmaxf(2.f * A, 0.f), 6.f);
  float x2 = 1.f;
  x[p*3+0] = x0; x[p*3+1] = x1; x[p*3+2] = x2;
  z[p*3+0] = x0; z[p*3+1] = x1; z[p*3+2] = x2;
  beta[p*3+0] = 0.f; beta[p*3+1] = 0.f; beta[p*3+2] = 0.f;
}

// ---------------- Gauss-Newton recon step (per pixel, analytic 3x3 inverse) --
__global__ void k_recon(const float* __restrict__ b, const float* __restrict__ tau,
                        const float* __restrict__ rho_p, float rho_mul,
                        const float* __restrict__ zin, const float* __restrict__ betain,
                        float* __restrict__ x, float* __restrict__ zout,
                        float* __restrict__ betaout, float* __restrict__ out_slab) {
  int p = blockIdx.x * blockDim.x + threadIdx.x;
  if (p >= NPIX) return;
  float rho = rho_p[0] * rho_mul;
  const float* bp = b + (size_t)p * NQ;
  float A  = x[p*3+0], Bv = x[p*3+1], R1 = x[p*3+2];
  float g0 = 0.f, g1 = 0.f, g2 = 0.f;
  float h00 = 0.f, h01 = 0.f, h02 = 0.f, h11 = 0.f, h12 = 0.f, h22 = 0.f;
  #pragma unroll
  for (int q = 0; q < NQ; ++q) {
    float t  = tau[q];
    float E  = expf(-R1 * t);
    float r  = (A - Bv * E) - bp[q];
    float jB = -E;
    float jR = Bv * t * E;
    g0 += r;  g1 += jB * r;  g2 += jR * r;
    h00 += 1.f;      h01 += jB;       h02 += jR;
    h11 += jB * jB;  h12 += jB * jR;  h22 += jR * jR;
  }
  const float inv8 = 1.f / (float)NQ;
  g0 *= inv8; g1 *= inv8; g2 *= inv8;
  g0 += rho * (A  + betain[p*3+0] - zin[p*3+0]);
  g1 += rho * (Bv + betain[p*3+1] - zin[p*3+1]);
  g2 += rho * (R1 + betain[p*3+2] - zin[p*3+2]);
  h00 += rho; h11 += rho; h22 += rho;
  float det = h00*(h11*h22 - h12*h12) - h01*(h01*h22 - h12*h02)
            + h02*(h01*h12 - h11*h02);
  if (det <= 0.f) {
    h00 += EPS_F; h11 += EPS_F; h22 += EPS_F;
    det = h00*(h11*h22 - h12*h12) - h01*(h01*h22 - h12*h02)
        + h02*(h01*h12 - h11*h02);
  }
  float id  = 1.f / det;
  float i00 = (h11*h22 - h12*h12) * id;
  float i01 = (h02*h12 - h01*h22) * id;
  float i02 = (h01*h12 - h02*h11) * id;
  float i11 = (h00*h22 - h02*h02) * id;
  float i12 = (h01*h02 - h00*h12) * id;
  float i22 = (h00*h11 - h01*h01) * id;
  float d0 = i00*g0 + i01*g1 + i02*g2;
  float d1 = i01*g0 + i11*g1 + i12*g2;
  float d2 = i02*g0 + i12*g1 + i22*g2;
  float n0 = fminf(fmaxf(A  - d0, 0.f), 3.f);
  float n1 = fminf(fmaxf(Bv - d1, 0.f), 6.f);
  float n2 = fminf(fmaxf(R1 - d2, 0.f), 50.f);
  x[p*3+0] = n0; x[p*3+1] = n1; x[p*3+2] = n2;
  if (zout)     { zout[p*3+0] = n0; zout[p*3+1] = n1; zout[p*3+2] = n2; }
  if (betaout)  { betaout[p*3+0] = 0.f; betaout[p*3+1] = 0.f; betaout[p*3+2] = 0.f; }
  if (out_slab) { out_slab[p*3+0] = n0; out_slab[p*3+1] = n1; out_slab[p*3+2] = n2; }
}

// ---------------- layer 0: 3 -> 64, VALU, ReLU, bf16 out ---------------------
__global__ void k_conv0(const float* __restrict__ zin, const float* __restrict__ w,
                        const float* __restrict__ bias, unsigned short* __restrict__ actout) {
  int id = blockIdx.x * blockDim.x + threadIdx.x;
  if (id >= NPIX * CH) return;
  int co = id & 63, p = id >> 6;
  int n = p / (IMG_H * IMG_W), rem = p % (IMG_H * IMG_W);
  int y = rem / IMG_W, xc = rem % IMG_W;
  float acc = bias[co];
  #pragma unroll
  for (int dy = -1; dy <= 1; ++dy)
    #pragma unroll
    for (int dx = -1; dx <= 1; ++dx) {
      int yy = y + dy, xx = xc + dx;
      if (yy < 0 || yy >= IMG_H || xx < 0 || xx >= IMG_W) continue;
      const float* zp = zin + (size_t)((n * IMG_H + yy) * IMG_W + xx) * 3;
      const float* wp = w + (size_t)(((dy + 1) * 3 + (dx + 1)) * 3) * CH + co;
      acc += zp[0]*wp[0] + zp[1]*wp[CH] + zp[2]*wp[2*CH];
    }
  actout[id] = f32_to_bf16(fmaxf(acc, 0.f));
}

// ---------------- layers 1..5: 64 -> 64 implicit-GEMM with bf16 WMMA ---------
// 128 threads = 4 waves; wave w owns N-tile w (16 out channels); each wave
// register-blocks 4 M-tiles (64 consecutive pixels; every 16-pixel tile lies
// inside one image row since 16 | 160). B fragments are reused across the 4
// M-tiles; A fragments are two aligned b128 loads from one neighbor pixel.
__global__ __launch_bounds__(128)
void k_conv_wmma(const unsigned short* __restrict__ actin,
                 const unsigned short* __restrict__ wpk,
                 const float* __restrict__ bias,
                 unsigned short* __restrict__ actout) {
  int lane  = threadIdx.x & 31;
  int wv    = threadIdx.x >> 5;                    // N-tile 0..3
  int pbase = blockIdx.x * (16 * MTILES);
  int n0    = wv * 16 + (lane & 15);
  int yv[MTILES], xv[MTILES], nv[MTILES];
  #pragma unroll
  for (int t = 0; t < MTILES; ++t) {
    int pa  = pbase + t * 16 + (lane & 15);        // pixel this lane gathers
    nv[t]   = pa / (IMG_H * IMG_W);
    int rem = pa % (IMG_H * IMG_W);
    yv[t]   = rem / IMG_W;
    xv[t]   = rem % IMG_W;
  }
  float bv = bias[n0];
  v8f acc[MTILES];
  #pragma unroll
  for (int t = 0; t < MTILES; ++t) acc[t] = (v8f){bv, bv, bv, bv, bv, bv, bv, bv};
  const ushort8 z8 = {0, 0, 0, 0, 0, 0, 0, 0};
  #pragma unroll
  for (int c = 0; c < KCHUNKS; ++c) {
    // A fragment layout: lanes 0-15 hold K {kb..kb+7, kb+16..kb+23}, kb=c*32;
    // lanes 16-31 same with kb += 8. Both 8-blocks land in one 3x3 tap.
    int kb  = c * 32 + ((lane >> 4) << 3);
    int tap = kb >> 6;
    int ci0 = kb & 63;
    int dy = tap / 3 - 1, dx = tap % 3 - 1;
    // B fragment: loaded once, reused by all MTILES WMMAs this chunk
    union { ushort8 h[2]; v16bf v; } bf;
    const ushort8* bp = (const ushort8*)(wpk +
        ((size_t)((c * 4 + wv) * 32 + lane)) * 16);
    bf.h[0] = bp[0]; bf.h[1] = bp[1];
    #pragma unroll
    for (int t = 0; t < MTILES; ++t) {
      int yy = yv[t] + dy, xx = xv[t] + dx;
      bool ok = (yy >= 0) & (yy < IMG_H) & (xx >= 0) & (xx < IMG_W);
      union { ushort8 h[2]; v16bf v; } af;
      if (ok) {
        const ushort8* ap = (const ushort8*)(actin +
            ((size_t)((nv[t] * IMG_H + yy) * IMG_W + xx) * CH + ci0));
        af.h[0] = ap[0];       // K = kb .. kb+7
        af.h[1] = ap[2];       // K = kb+16 .. kb+23  (+16 channels)
      } else {
        af.h[0] = z8; af.h[1] = z8;
      }
      acc[t] = __builtin_amdgcn_wmma_f32_16x16x32_bf16(
          false, af.v, false, bf.v, (short)0, acc[t], false, false);
    }
  }
  // D layout: lane L, acc[t][v] -> M = v + 8*(L>=16) (+ tile offs), N = L&15
  #pragma unroll
  for (int t = 0; t < MTILES; ++t) {
    int mbase = pbase + t * 16 + ((lane >> 4) << 3);
    #pragma unroll
    for (int v = 0; v < 8; ++v) {
      float f = fmaxf(acc[t][v], 0.f);              // fused ReLU
      actout[(size_t)(mbase + v) * CH + n0] = f32_to_bf16(f);
    }
  }
}

// ---------------- layer 6: 64 -> 3, VALU, no ReLU, f32 out -------------------
__global__ void k_conv6(const unsigned short* __restrict__ actin,
                        const float* __restrict__ w, const float* __restrict__ bias,
                        float* __restrict__ fz, float* __restrict__ out_slab) {
  int id = blockIdx.x * blockDim.x + threadIdx.x;
  if (id >= NPIX * 3) return;
  int co = id % 3, p = id / 3;
  int n = p / (IMG_H * IMG_W), rem = p % (IMG_H * IMG_W);
  int y = rem / IMG_W, xc = rem % IMG_W;
  float acc = bias[co];
  for (int dy = -1; dy <= 1; ++dy)
    for (int dx = -1; dx <= 1; ++dx) {
      int yy = y + dy, xx = xc + dx;
      if (yy < 0 || yy >= IMG_H || xx < 0 || xx >= IMG_W) continue;
      const unsigned short* ap = actin + (size_t)((n * IMG_H + yy) * IMG_W + xx) * CH;
      const float* wp = w + (size_t)(((dy + 1) * 3 + (dx + 1)) * CH) * 3 + co;
      #pragma unroll 8
      for (int ci = 0; ci < CH; ++ci) acc += bf16_to_f32(ap[ci]) * wp[ci * 3];
    }
  fz[id] = acc;
  out_slab[id] = acc;
}

// ---------------- z / beta update + output slabs -----------------------------
__global__ void k_zup(const float* __restrict__ x, const float* __restrict__ fz,
                      float* __restrict__ z, float* __restrict__ beta,
                      const float* __restrict__ lam_p, const float* __restrict__ rho_p,
                      float* __restrict__ out_z) {
  int id = blockIdx.x * blockDim.x + threadIdx.x;
  if (id >= NPIX * 3) return;
  int c = id % 3;
  float rho = rho_p[0], lam = lam_p[0];
  float den = rho + lam;
  float num = rho * (x[id] + beta[id]) + lam * fz[id];
  float zn  = (den == 0.f) ? 0.f : num / den;
  zn = fminf(fmaxf(zn, 0.f), chmax(c));
  float bn = beta[id] + x[id] - zn;
  z[id] = zn; beta[id] = bn; out_z[id] = zn;
}

__global__ void k_copy(const float* __restrict__ src, float* __restrict__ dst) {
  int id = blockIdx.x * blockDim.x + threadIdx.x;
  if (id < NPIX * 3) dst[id] = src[id];
}

// ---------------------------------------------------------------------------
extern "C" void kernel_launch(void* const* d_in, const int* in_sizes, int n_in,
                              void* d_out, int out_size, void* d_ws, size_t ws_size,
                              hipStream_t stream) {
  (void)in_sizes; (void)n_in; (void)out_size; (void)ws_size;
  const float* b    = (const float*)d_in[0];
  const float* tau  = (const float*)d_in[1];
  const float* lam  = (const float*)d_in[2];
  const float* rho  = (const float*)d_in[3];
  const float* dw0  = (const float*)d_in[4];
  const float* dwL[5] = {(const float*)d_in[5], (const float*)d_in[6],
                         (const float*)d_in[7], (const float*)d_in[8],
                         (const float*)d_in[9]};
  const float* dw6  = (const float*)d_in[10];
  const float* db0  = (const float*)d_in[11];
  const float* dbL[5] = {(const float*)d_in[12], (const float*)d_in[13],
                         (const float*)d_in[14], (const float*)d_in[15],
                         (const float*)d_in[16]};
  const float* db6  = (const float*)d_in[17];
  float* out = (float*)d_out;

  char* ws = (char*)d_ws;
  float*          x    = (float*)(ws + 0);
  float*          z    = (float*)(ws + 1228800);
  float*          beta = (float*)(ws + 2457600);
  float*          fz   = (float*)(ws + 3686400);
  unsigned short* actA = (unsigned short*)(ws + 4915200);
  unsigned short* actB = (unsigned short*)(ws + 18022400);
  unsigned short* wpk  = (unsigned short*)(ws + 31129600);

  const int TPB = 256;
  const size_t SLAB = (size_t)NPIX * 3;
  const int WMMA_BLOCKS = NPIX / (16 * MTILES);   // 1600

  // one-time (per launch, deterministic) bf16 weight repack for the WMMA path
  k_repack<<<720, TPB, 0, stream>>>(dwL[0], dwL[1], dwL[2], dwL[3], dwL[4], wpk);

  // x init, z = x, beta = 0
  k_init<<<400, TPB, 0, stream>>>(b, x, z, beta);

  // first recon block with rho = 0, then z = x, beta = 0
  k_recon<<<400, TPB, 0, stream>>>(b, tau, rho, 0.f, z, beta, x, z, beta, nullptr);

  for (int s = 0; s < 6; ++s) {
    // x-update (rho active); append x  -> slab 3s
    k_recon<<<400, TPB, 0, stream>>>(b, tau, rho, 1.f, z, beta, x,
                                     nullptr, nullptr, out + (size_t)(3*s) * SLAB);
    // CNN on z (the z from before this iteration's update)
    k_conv0<<<25600, TPB, 0, stream>>>(z, dw0, db0, actA);
    k_conv_wmma<<<WMMA_BLOCKS, 128, 0, stream>>>(actA, wpk + 0*WPK_PER_LAYER, dbL[0], actB);
    k_conv_wmma<<<WMMA_BLOCKS, 128, 0, stream>>>(actB, wpk + 1*WPK_PER_LAYER, dbL[1], actA);
    k_conv_wmma<<<WMMA_BLOCKS, 128, 0, stream>>>(actA, wpk + 2*WPK_PER_LAYER, dbL[2], actB);
    k_conv_wmma<<<WMMA_BLOCKS, 128, 0, stream>>>(actB, wpk + 3*WPK_PER_LAYER, dbL[3], actA);
    k_conv_wmma<<<WMMA_BLOCKS, 128, 0, stream>>>(actA, wpk + 4*WPK_PER_LAYER, dbL[4], actB);
    // append fz -> slab 3s+1
    k_conv6<<<1200, TPB, 0, stream>>>(actB, dw6, db6, fz,
                                      out + (size_t)(3*s + 1) * SLAB);
    // z-update + beta-update; append z -> slab 3s+2
    k_zup<<<1200, TPB, 0, stream>>>(x, fz, z, beta, lam, rho,
                                    out + (size_t)(3*s + 2) * SLAB);
  }
  // final x -> slab 18
  k_copy<<<1200, TPB, 0, stream>>>(x, out + (size_t)18 * SLAB);
}
